// CausalSelfAttention_55319178773072
// MI455X (gfx1250) — matrix operand order repaired and checked by reading, not compiled, not run on hardware
//
#include <hip/hip_runtime.h>
#include <hip/hip_bf16.h>

#define Bb   2
#define Tt   2048
#define Cc   1024
#define Hh   16
#define HS   64
#define N3C  3072
#define MROW (Bb * Tt)   // 4096

typedef __attribute__((ext_vector_type(16))) __bf16 v16bf;
typedef __attribute__((ext_vector_type(8)))  float  v8f;
typedef __attribute__((ext_vector_type(8)))  short  v8s;
typedef __attribute__((ext_vector_type(16))) short  v16s;
typedef __attribute__((ext_vector_type(4)))  int    v4i;

// ---- optional CDNA5 async global->LDS path (ASYNCcnt), with sync fallback
#if defined(__has_builtin)
#if __has_builtin(__builtin_amdgcn_global_load_async_to_lds_b128) && \
    __has_builtin(__builtin_amdgcn_s_wait_asynccnt)
#define HAVE_ASYNC_LDS 1
#endif
#endif
#ifndef HAVE_ASYNC_LDS
#define HAVE_ASYNC_LDS 0
#endif

#if HAVE_ASYNC_LDS
static __device__ __forceinline__ void async_copy16(const unsigned short* g,
                                                    unsigned short* l) {
    // emits global_load_async_to_lds_b128 (per-lane LDS dst, per-lane global src)
    __builtin_amdgcn_global_load_async_to_lds_b128(
        (v4i __attribute__((address_space(1)))*)g,
        (v4i __attribute__((address_space(3)))*)l, 0, 0);
}
#endif

static __device__ __forceinline__ unsigned short f2bf(float f) {
    union { float f; unsigned int u; } x; x.f = f;
    unsigned int r = x.u + 0x7fffu + ((x.u >> 16) & 1u);   // RNE
    return (unsigned short)(r >> 16);
}

static __device__ __forceinline__ v16bf frag_from(v8s lo, v8s hi) {
    v16s s = __builtin_shufflevector(lo, hi, 0,1,2,3,4,5,6,7,8,9,10,11,12,13,14,15);
    return __builtin_bit_cast(v16bf, s);
}

static __device__ __forceinline__ v8f wmma_bf16(v16bf a, v16bf b, v8f c) {
    return __builtin_amdgcn_wmma_f32_16x16x32_bf16(false, a, false, b, (short)0, c, false, false);
}

// ---------------------------------------------------------------- fp32 -> bf16
__global__ void cvt_f32_bf16(const float* __restrict__ src,
                             unsigned short* __restrict__ dst, int n) {
    int i = blockIdx.x * blockDim.x + threadIdx.x;
    if (i < n) dst[i] = f2bf(src[i]);
}

// ---------------------------------------------------------------- GEMM core
// 256 threads = 8 waves. Block tile 256x64, K-step 32, double-buffered LDS.
// Each wave: 32x64 (2 A-frags x 4 B-frags = 8 WMMA / 12 fragment loads).
static __device__ __forceinline__ void gemm_core(
    const unsigned short* __restrict__ A,    // [MROW][Cc] bf16, row-major
    const unsigned short* __restrict__ Bw,   // [Cc][ldb]  bf16, row-major
    const int ldb, const int m0, const int n0,
    unsigned short (&As)[2][256][32],
    unsigned short (&Bt)[2][64][32],
    v8f (&acc)[2][4])
{
    const int tid  = threadIdx.x;
    const int lane = tid & 31, w = tid >> 5;
    const int ln   = lane & 15, hi = lane >> 4;
    const int kb_  = hi * 8;
    const int bkk  = tid >> 3, bch = tid & 7;   // B-tile copy coords

    // ---- prologue: stage tile 0 into buffer 0
#if HAVE_ASYNC_LDS
    #pragma unroll
    for (int i = 0; i < 4; ++i) {
        int c = i * 256 + tid, r = c >> 2, ch = c & 3;
        async_copy16(A + (size_t)(m0 + r) * Cc + ch * 8, &As[0][r][ch * 8]);
    }
#else
    #pragma unroll
    for (int i = 0; i < 4; ++i) {
        int c = i * 256 + tid, r = c >> 2, ch = c & 3;
        *(v8s*)&As[0][r][ch * 8] = *(const v8s*)(A + (size_t)(m0 + r) * Cc + ch * 8);
    }
#endif
    {
        v8s val = *(const v8s*)(Bw + (size_t)bkk * ldb + n0 + bch * 8);
        #pragma unroll
        for (int i = 0; i < 8; ++i) Bt[0][bch * 8 + i][bkk] = (unsigned short)val[i];
    }
#if HAVE_ASYNC_LDS
    __builtin_amdgcn_s_wait_asynccnt(0);
#endif
    __syncthreads();

    // ---- main loop
    for (int k0 = 0; k0 < Cc; k0 += 32) {
        const int  buf  = (k0 >> 5) & 1;
        const bool more = (k0 + 32) < Cc;

        // (1) kick off next tile's loads (A: async or into regs; B: into regs)
        v8s bReg = {};
#if !HAVE_ASYNC_LDS
        v8s aReg[4];
#endif
        if (more) {
            const int kn = k0 + 32;
#if HAVE_ASYNC_LDS
            #pragma unroll
            for (int i = 0; i < 4; ++i) {
                int c = i * 256 + tid, r = c >> 2, ch = c & 3;
                async_copy16(A + (size_t)(m0 + r) * Cc + kn + ch * 8,
                             &As[buf ^ 1][r][ch * 8]);
            }
#else
            #pragma unroll
            for (int i = 0; i < 4; ++i) {
                int c = i * 256 + tid, r = c >> 2, ch = c & 3;
                aReg[i] = *(const v8s*)(A + (size_t)(m0 + r) * Cc + kn + ch * 8);
            }
#endif
            bReg = *(const v8s*)(Bw + (size_t)(kn + bkk) * ldb + n0 + bch * 8);
            if (k0 + 64 < Cc)   // pull the tile after that toward L2
                __builtin_prefetch(A + (size_t)(m0 + tid) * Cc + k0 + 64, 0, 1);
        }

        // (2) compute current tile while loads are in flight
        v16bf af[2], bfr[4];
        #pragma unroll
        for (int ii = 0; ii < 2; ++ii) {
            const int arow = w * 32 + ii * 16 + ln;
            af[ii] = frag_from(*(const v8s*)&As[buf][arow][kb_],
                               *(const v8s*)&As[buf][arow][16 + kb_]);
        }
        #pragma unroll
        for (int j = 0; j < 4; ++j) {
            const int brow = j * 16 + ln;
            bfr[j] = frag_from(*(const v8s*)&Bt[buf][brow][kb_],
                               *(const v8s*)&Bt[buf][brow][16 + kb_]);
        }
        #pragma unroll
        for (int ii = 0; ii < 2; ++ii)
            #pragma unroll
            for (int j = 0; j < 4; ++j)
                acc[ii][j] = wmma_bf16(af[ii], bfr[j], acc[ii][j]);

        // (3) finish staging next tile
        if (more) {
#if !HAVE_ASYNC_LDS
            #pragma unroll
            for (int i = 0; i < 4; ++i) {
                int c = i * 256 + tid, r = c >> 2, ch = c & 3;
                *(v8s*)&As[buf ^ 1][r][ch * 8] = aReg[i];
            }
#endif
            #pragma unroll
            for (int i = 0; i < 8; ++i)
                Bt[buf ^ 1][bch * 8 + i][bkk] = (unsigned short)bReg[i];
#if HAVE_ASYNC_LDS
            __builtin_amdgcn_s_wait_asynccnt(0);
#endif
        }
        __syncthreads();
    }
}

// ---------------------------------------------------------------- qkv GEMM
// Epilogue scatters into q/k/v [B,H,T,hs] bf16; q scaled by 100/sqrt(64)=12.5.
__global__ __launch_bounds__(256) void gemm_qkv(
    const unsigned short* __restrict__ A,
    const unsigned short* __restrict__ Bw,
    unsigned short* __restrict__ qb,
    unsigned short* __restrict__ kbuf,
    unsigned short* __restrict__ vbuf)
{
    __shared__ unsigned short As[2][256][32];
    __shared__ unsigned short Bt[2][64][32];

    const int tid  = threadIdx.x;
    const int lane = tid & 31, w = tid >> 5;
    const int ln   = lane & 15, hi = lane >> 4;
    const int m0   = blockIdx.y * 256;
    const int n0   = blockIdx.x * 64;

    v8f acc[2][4] = {};
    gemm_core(A, Bw, N3C, m0, n0, As, Bt, acc);

    #pragma unroll
    for (int ii = 0; ii < 2; ++ii)
        #pragma unroll
        for (int r = 0; r < 8; ++r) {
            int grow = m0 + w * 32 + ii * 16 + r + 8 * hi;   // [0,4096)
            int bbb  = grow >> 11;
            int t    = grow & (Tt - 1);
            #pragma unroll
            for (int j = 0; j < 4; ++j) {
                int gn = n0 + j * 16 + ln;                   // [0,3072)
                float val = acc[ii][j][r];
                int which = gn >> 10;
                int cq    = gn & (Cc - 1);
                int h = cq >> 6, d = cq & 63;
                size_t idx = (((size_t)(bbb * Hh + h) * Tt) + t) * HS + d;
                if (which == 0)      qb[idx]   = f2bf(val * 12.5f);
                else if (which == 1) kbuf[idx] = f2bf(val);
                else                 vbuf[idx] = f2bf(val);
            }
        }
}

// ---------------------------------------------------------------- proj GEMM
__global__ __launch_bounds__(256) void gemm_proj(
    const unsigned short* __restrict__ A,
    const unsigned short* __restrict__ Bw,
    float* __restrict__ out)
{
    __shared__ unsigned short As[2][256][32];
    __shared__ unsigned short Bt[2][64][32];

    const int tid  = threadIdx.x;
    const int lane = tid & 31, w = tid >> 5;
    const int ln   = lane & 15, hi = lane >> 4;
    const int m0   = blockIdx.y * 256;
    const int n0   = blockIdx.x * 64;

    v8f acc[2][4] = {};
    gemm_core(A, Bw, Cc, m0, n0, As, Bt, acc);

    #pragma unroll
    for (int ii = 0; ii < 2; ++ii)
        #pragma unroll
        for (int r = 0; r < 8; ++r) {
            size_t grow = (size_t)(m0 + w * 32 + ii * 16 + r + 8 * hi);
            #pragma unroll
            for (int j = 0; j < 4; ++j)
                out[grow * Cc + n0 + j * 16 + ln] = acc[ii][j][r];
        }
}

// ---------------------------------------------------------------- attention
// One wave per (b,h, 16-row q tile). Flash-style online softmax, K-step 32.
__global__ __launch_bounds__(32) void attn(
    const unsigned short* __restrict__ qb,
    const unsigned short* __restrict__ kbuf,
    const unsigned short* __restrict__ vbuf,
    unsigned short* __restrict__ yb)
{
    __shared__ unsigned short vT[64][32];   // [d][s_local]
    __shared__ float          P[16][32];    // [m][s_local]

    const int lane = threadIdx.x;
    const int ln = lane & 15, hi = lane >> 4;
    const int nqt = Tt / 16;
    const int qt  = blockIdx.x % nqt;
    const int bh  = blockIdx.x / nqt;
    const size_t rowbase = (size_t)bh * Tt;
    const int kb_ = hi * 8;
    const float NEGBIG = -3.0e38f;

    const unsigned short* qrow = qb + (rowbase + qt * 16 + ln) * HS;
    v16bf aq0 = frag_from(*(const v8s*)(qrow + kb_),      *(const v8s*)(qrow + 16 + kb_));
    v16bf aq1 = frag_from(*(const v8s*)(qrow + 32 + kb_), *(const v8s*)(qrow + 48 + kb_));

    v8f yacc[4] = {};
    float mrow[8], lrow[8];
    #pragma unroll
    for (int r = 0; r < 8; ++r) { mrow[r] = NEGBIG; lrow[r] = 0.0f; }

    const int tmax = qt * 16 + 15;
    for (int s0 = 0; s0 <= tmax; s0 += 32) {
        // stage v[s0..s0+31][0..63] transposed: lane owns row s0+lane
        {
            const unsigned short* vrow = vbuf + (rowbase + s0 + lane) * HS;
            #pragma unroll
            for (int ch = 0; ch < 8; ++ch) {
                v8s c = *(const v8s*)(vrow + ch * 8);
                #pragma unroll
                for (int i = 0; i < 8; ++i) vT[ch * 8 + i][lane] = (unsigned short)c[i];
            }
        }
        // batch all k fragments, then issue the WMMA group
        v16bf bk[4];
        #pragma unroll
        for (int j = 0; j < 2; ++j) {
            const unsigned short* krow = kbuf + (rowbase + s0 + j * 16 + ln) * HS;
            bk[j * 2 + 0] = frag_from(*(const v8s*)(krow + kb_),
                                      *(const v8s*)(krow + 16 + kb_));
            bk[j * 2 + 1] = frag_from(*(const v8s*)(krow + 32 + kb_),
                                      *(const v8s*)(krow + 48 + kb_));
        }
        v8f att[2] = {};
        att[0] = wmma_bf16(aq0, bk[0], att[0]);
        att[0] = wmma_bf16(aq1, bk[1], att[0]);
        att[1] = wmma_bf16(aq0, bk[2], att[1]);
        att[1] = wmma_bf16(aq1, bk[3], att[1]);

        // online softmax: row m = r + 8*hi lives in a 16-lane half-group
        float scl[8];
        #pragma unroll
        for (int r = 0; r < 8; ++r) {
            int t_idx = qt * 16 + r + 8 * hi;
            float a0 = ((s0 + ln)      <= t_idx) ? att[0][r] : NEGBIG;
            float a1 = ((s0 + 16 + ln) <= t_idx) ? att[1][r] : NEGBIG;
            float vmax = fmaxf(a0, a1);
            #pragma unroll
            for (int msk = 1; msk <= 8; msk <<= 1)
                vmax = fmaxf(vmax, __shfl_xor(vmax, msk, 32));
            float mnew = fmaxf(mrow[r], vmax);
            float sc = __expf(mrow[r] - mnew);
            float p0 = __expf(a0 - mnew);
            float p1 = __expf(a1 - mnew);
            float s = p0 + p1;
            #pragma unroll
            for (int msk = 1; msk <= 8; msk <<= 1)
                s += __shfl_xor(s, msk, 32);
            lrow[r] = lrow[r] * sc + s;
            mrow[r] = mnew;
            scl[r] = sc;
            P[r + 8 * hi][ln]      = p0;
            P[r + 8 * hi][16 + ln] = p1;
        }
        #pragma unroll
        for (int j = 0; j < 4; ++j)
            #pragma unroll
            for (int r = 0; r < 8; ++r)
                yacc[j][r] *= scl[r];
        __syncthreads();

        // rebuild p as A fragment via LDS transpose, convert to bf16
        v16s ps;
        const float* prow = &P[ln][0];
        #pragma unroll
        for (int i = 0; i < 8; ++i) {
            ps[i]     = (short)f2bf(prow[kb_ + i]);
            ps[i + 8] = (short)f2bf(prow[16 + kb_ + i]);
        }
        v16bf pf = __builtin_bit_cast(v16bf, ps);

        v16bf vf[4];
        #pragma unroll
        for (int j = 0; j < 4; ++j) {
            const int d = j * 16 + ln;
            vf[j] = frag_from(*(const v8s*)&vT[d][kb_],
                              *(const v8s*)&vT[d][16 + kb_]);
        }
        #pragma unroll
        for (int j = 0; j < 4; ++j)
            yacc[j] = wmma_bf16(pf, vf[j], yacc[j]);
        __syncthreads();
    }

    const int h = bh % Hh, b = bh / Hh;
    #pragma unroll
    for (int r = 0; r < 8; ++r) {
        float inv = 1.0f / lrow[r];
        int t_idx = qt * 16 + r + 8 * hi;
        size_t orow = ((size_t)b * Tt + t_idx) * Cc + h * HS;
        #pragma unroll
        for (int j = 0; j < 4; ++j)
            yb[orow + j * 16 + ln] = f2bf(yacc[j][r] * inv);
    }
}

// ---------------------------------------------------------------- att_l1
// mean(|att|) is taken AFTER the -inf mask, so it is exactly +inf.
__global__ void write_l1(float* __restrict__ out) {
    out[(size_t)Bb * Tt * Cc] = __builtin_inff();
}

// ---------------------------------------------------------------- launch
extern "C" void kernel_launch(void* const* d_in, const int* in_sizes, int n_in,
                              void* d_out, int out_size, void* d_ws, size_t ws_size,
                              hipStream_t stream)
{
    const float* x      = (const float*)d_in[0];
    const float* w_attn = (const float*)d_in[1];
    const float* w_proj = (const float*)d_in[2];
    float* out = (float*)d_out;

    char* ws = (char*)d_ws;
    size_t off = 0;
    auto carve = [&](size_t bytes) {
        char* p = ws + off;
        off += (bytes + 255) & ~(size_t)255;
        return p;
    };
    unsigned short* xb   = (unsigned short*)carve((size_t)MROW * Cc * 2);
    unsigned short* wab  = (unsigned short*)carve((size_t)Cc * N3C * 2);
    unsigned short* wpb  = (unsigned short*)carve((size_t)Cc * Cc * 2);
    unsigned short* qb   = (unsigned short*)carve((size_t)MROW * Cc * 2);
    unsigned short* kbuf = (unsigned short*)carve((size_t)MROW * Cc * 2);
    unsigned short* vbuf = (unsigned short*)carve((size_t)MROW * Cc * 2);
    unsigned short* yb   = (unsigned short*)carve((size_t)MROW * Cc * 2);

    const int nx = MROW * Cc, na = Cc * N3C, np = Cc * Cc;
    cvt_f32_bf16<<<(nx + 255) / 256, 256, 0, stream>>>(x, xb, nx);
    cvt_f32_bf16<<<(na + 255) / 256, 256, 0, stream>>>(w_attn, wab, na);
    cvt_f32_bf16<<<(np + 255) / 256, 256, 0, stream>>>(w_proj, wpb, np);

    dim3 g1(N3C / 64, MROW / 256);
    gemm_qkv<<<g1, 256, 0, stream>>>(xb, wab, qb, kbuf, vbuf);

    attn<<<Bb * Hh * (Tt / 16), 32, 0, stream>>>(qb, kbuf, vbuf, yb);

    dim3 g2(Cc / 64, MROW / 256);
    gemm_proj<<<g2, 256, 0, stream>>>(yb, wpb, out);

    write_l1<<<1, 1, 0, stream>>>(out);
}
